// TransformerEncoderLayer_73598559584861
// MI455X (gfx1250) — matrix operand order repaired and checked by reading, not compiled, not run on hardware
//
#include <hip/hip_runtime.h>
#include <math.h>

// ---------------------------------------------------------------------------
// Transformer encoder layer for gfx1250 (MI455X): bf16 WMMA everywhere,
// TDM (tensor_load_to_lds) double-buffered tile DMA in the GEMMs.
// ---------------------------------------------------------------------------

#define DM      1024
#define DFF     4096
#define SEQLEN  2048
#define NBATCH  2
#define NHEADS  16
#define DHEAD   64
#define ROWS    (NBATCH * SEQLEN)   // 4096 token rows

typedef __attribute__((ext_vector_type(16))) __bf16 bf16x16;
typedef __attribute__((ext_vector_type(8)))  __bf16 bf16x8;
typedef __attribute__((ext_vector_type(8)))  float  f32x8;
typedef __attribute__((ext_vector_type(4)))  unsigned int u32x4;
typedef __attribute__((ext_vector_type(8)))  int i32x8;
typedef __attribute__((ext_vector_type(4)))  int i32x4;

#define USE_TDM (__has_builtin(__builtin_amdgcn_tensor_load_to_lds) && \
                 __has_builtin(__builtin_amdgcn_s_wait_tensorcnt))

__device__ __forceinline__ f32x8 vzero8() {
  f32x8 z;
  #pragma unroll
  for (int i = 0; i < 8; ++i) z[i] = 0.0f;
  return z;
}

__device__ __forceinline__ f32x8 wmma_bf16(bf16x16 a, bf16x16 b, f32x8 c) {
  // D = A(16x32 bf16) x B(32x16 bf16) + C(16x16 f32) -> v_wmma_f32_16x16x32_bf16
  return __builtin_amdgcn_wmma_f32_16x16x32_bf16(false, a, false, b, (short)0, c,
                                                 false, false);
}

#define SHUF16(lo, hi) \
  __builtin_shufflevector(lo, hi, 0,1,2,3,4,5,6,7,8,9,10,11,12,13,14,15)

// A-matrix 16x32 bf16 fragment: lane m = lane&15, half = lane>>4.
// VGPR0..3 hold K = half*8 + {0..7}, VGPR4..7 hold K = 16 + half*8 + {0..7}.
__device__ __forceinline__ bf16x16 frag_a_ld(const __bf16* p, int half) {
  bf16x8 lo = *(const bf16x8*)(p + half * 8);
  bf16x8 hi = *(const bf16x8*)(p + 16 + half * 8);
  return SHUF16(lo, hi);
}

// B-matrix 32x16 bf16 fragment: lane n = lane&15; K = half*8 + {0..7}, 16 + half*8 + {0..7}.
__device__ __forceinline__ bf16x16 frag_b_ld(const __bf16* p, int half) {
  bf16x8 lo = *(const bf16x8*)(p + half * 8);
  bf16x8 hi = *(const bf16x8*)(p + 16 + half * 8);
  return SHUF16(lo, hi);
}

#if USE_TDM
// --- Tensor Data Mover helpers (D# per CDNA5 ISA) ----
__device__ __forceinline__ unsigned lds_lo32(const void* p) {
  // generic LDS pointer: low 32 bits are the LDS byte address
  return (unsigned)(unsigned long long)p;
}

// D# group 0: count=1 | lds_addr | global_addr[56:0] | type=2
__device__ __forceinline__ u32x4 tdm_g0(unsigned lds_addr, const void* gaddr) {
  const unsigned long long ga = (unsigned long long)gaddr;
  u32x4 g0;
  g0.x = 1u;                                                 // count=1 (valid)
  g0.y = lds_addr;
  g0.z = (unsigned)(ga & 0xFFFFFFFFu);                       // addr[31:0]
  g0.w = (unsigned)((ga >> 32) & 0x1FFFFFFu) | (2u << 30);   // addr[56:32]|type=2
  return g0;
}

// D# group 1 for a 2D bf16 tile: tile_k x tile_rows out of a rows x k_len
// tensor with row stride `stride` (elements). LDS pad: 16B after every 64B
// stored  (pad_interval code 3 = 16 DWORDs, pad_amount code 3 = 4 DWORDs)
// -> padded LDS row stride = 40 bf16 elements for tile_k = 32.
__device__ __forceinline__ i32x8 tdm_g1_2d(unsigned k_len, unsigned rows,
                                           unsigned tile_k, unsigned tile_rows,
                                           unsigned stride) {
  i32x8 g;
  g[0] = (int)((1u << 16) | (1u << 20) | (3u << 22) | (3u << 25));
  g[1] = (int)((k_len & 0xFFFFu) << 16);                        // dim0[15:0]
  g[2] = (int)(((k_len >> 16) & 0xFFFFu) | ((rows & 0xFFFFu) << 16));
  g[3] = (int)(((rows >> 16) & 0xFFFFu) | ((tile_k & 0xFFFFu) << 16));
  g[4] = (int)(tile_rows & 0xFFFFu);                            // tile_dim1
  g[5] = (int)stride;                                           // dim0_stride
  g[6] = 0;
  g[7] = 0;
  return g;
}

__device__ __forceinline__ void tdm_load(u32x4 g0, i32x8 g1) {
  const i32x4 z4 = {0, 0, 0, 0};
#if defined(__clang_major__) && (__clang_major__ >= 23)
  const i32x8 z8 = {0, 0, 0, 0, 0, 0, 0, 0};
  __builtin_amdgcn_tensor_load_to_lds(g0, g1, z4, z4, z8, 0);
#else
  __builtin_amdgcn_tensor_load_to_lds(g0, g1, z4, z4, 0);
#endif
}
#endif  // USE_TDM

// ---------------------------------------------------------------------------
// f32 -> bf16 bulk convert (weights)
// ---------------------------------------------------------------------------
__global__ __launch_bounds__(256) void cvt_f32_bf16(const float* __restrict__ in,
                                                    __bf16* __restrict__ out, int n) {
  int i = blockIdx.x * 256 + threadIdx.x;
  if (i < n) out[i] = (__bf16)in[i];
}

// ---------------------------------------------------------------------------
// LayerNorm (f32 in) -> bf16 out. One 256-thread block (8 waves) per row.
// ---------------------------------------------------------------------------
__global__ __launch_bounds__(256) void layernorm_to_bf16(
    const float* __restrict__ in, const float* __restrict__ gamma,
    const float* __restrict__ beta, __bf16* __restrict__ out) {
  const int row = blockIdx.x;
  const float* x = in + (size_t)row * DM;
  __shared__ float rsum[8], rsq[8];
  float s = 0.f, s2 = 0.f;
  for (int i = threadIdx.x; i < DM; i += 256) {
    float v = x[i];
    s += v;
    s2 += v * v;
  }
  #pragma unroll
  for (int off = 16; off >= 1; off >>= 1) {
    s  += __shfl_xor(s, off, 32);
    s2 += __shfl_xor(s2, off, 32);
  }
  if ((threadIdx.x & 31) == 0) {
    rsum[threadIdx.x >> 5] = s;
    rsq[threadIdx.x >> 5]  = s2;
  }
  __syncthreads();
  float ts = 0.f, t2 = 0.f;
  #pragma unroll
  for (int i = 0; i < 8; ++i) { ts += rsum[i]; t2 += rsq[i]; }
  const float mu  = ts * (1.0f / DM);
  const float var = t2 * (1.0f / DM) - mu * mu;
  const float inv = rsqrtf(var + 1e-5f);
  for (int i = threadIdx.x; i < DM; i += 256) {
    float v = (x[i] - mu) * inv * gamma[i] + beta[i];
    out[(size_t)row * DM + i] = (__bf16)v;
  }
}

// ---------------------------------------------------------------------------
// Tiled WMMA GEMM:  C[M,N] = A[M,K] * B[N,K]^T   (A,B bf16; acc f32)
// Block = 256 threads (8 waves), tile 128x128, k-step 32.
// TDM path: wave 0 issues tensor_load_to_lds for the NEXT k-tile (A and B)
// into the alternate LDS buffer, then s_wait_tensorcnt for the current one;
// DMA overlaps the 8 WMMAs/wave of the current tile.
// EPI: 0 = store bf16, 1 = exact GELU then store bf16, 2 = +resid store f32
// ---------------------------------------------------------------------------
template <int EPI>
__global__ __launch_bounds__(256) void gemm_bf16_nt(
    const __bf16* __restrict__ A, const __bf16* __restrict__ B,
    const float* __restrict__ resid, void* __restrict__ Cout,
    int M, int N, int K) {
  const int tid  = threadIdx.x;
  const int lane = tid & 31;
  const int wave = tid >> 5;
  const int wm   = wave >> 2;      // 0..1  (64 M-rows each)
  const int wn   = wave & 3;       // 0..3  (32 N-cols each)
  const int half = lane >> 4;
  const int nl   = lane & 15;
  const int bm   = blockIdx.x * 128;
  const int bn   = blockIdx.y * 128;

  f32x8 acc[4][2];
  #pragma unroll
  for (int i = 0; i < 4; ++i)
    #pragma unroll
    for (int j = 0; j < 2; ++j) acc[i][j] = vzero8();

#if USE_TDM
  __shared__ __bf16 As[2][128][40];   // 40 = 32 k + TDM pad (16B per 64B)
  __shared__ __bf16 Bs[2][128][40];
  const int nt = K >> 5;
  const i32x8 g1A = tdm_g1_2d((unsigned)K, (unsigned)M, 32u, 128u, (unsigned)K);
  const i32x8 g1B = tdm_g1_2d((unsigned)K, (unsigned)N, 32u, 128u, (unsigned)K);
  const unsigned ldsA[2] = { lds_lo32(&As[0][0][0]), lds_lo32(&As[1][0][0]) };
  const unsigned ldsB[2] = { lds_lo32(&Bs[0][0][0]), lds_lo32(&Bs[1][0][0]) };
  const __bf16* Abase = A + (size_t)bm * K;
  const __bf16* Bbase = B + (size_t)bn * K;

  if (wave == 0) {                       // prime the pipeline: tile 0
    tdm_load(tdm_g0(ldsA[0], Abase), g1A);
    tdm_load(tdm_g0(ldsB[0], Bbase), g1B);
  }
  for (int it = 0; it < nt; ++it) {
    const int cur = it & 1;
    if (wave == 0) {
      if (it + 1 < nt) {                 // issue next tile into other buffer
        tdm_load(tdm_g0(ldsA[cur ^ 1], Abase + (size_t)(it + 1) * 32), g1A);
        tdm_load(tdm_g0(ldsB[cur ^ 1], Bbase + (size_t)(it + 1) * 32), g1B);
        __builtin_amdgcn_s_wait_tensorcnt(2);   // current tile (2 ops) done
      } else {
        __builtin_amdgcn_s_wait_tensorcnt(0);
      }
    }
    __syncthreads();
    bf16x16 af[4], bfr[2];
    #pragma unroll
    for (int i = 0; i < 4; ++i)
      af[i] = frag_a_ld(&As[cur][wm * 64 + i * 16 + nl][0], half);
    #pragma unroll
    for (int j = 0; j < 2; ++j)
      bfr[j] = frag_b_ld(&Bs[cur][wn * 32 + j * 16 + nl][0], half);
    #pragma unroll
    for (int i = 0; i < 4; ++i)
      #pragma unroll
      for (int j = 0; j < 2; ++j) acc[i][j] = wmma_bf16(af[i], bfr[j], acc[i][j]);
    __syncthreads();                     // all done reading As/Bs[cur]
  }
#else
  __shared__ __bf16 As[128][40];
  __shared__ __bf16 Bs[128][40];
  for (int k0 = 0; k0 < K; k0 += 32) {
    __syncthreads();
    #pragma unroll
    for (int i = 0; i < 2; ++i) {        // 512 16B chunks, 2 per thread
      const int c   = tid + 256 * i;
      const int r   = c >> 2;
      const int col = (c & 3) * 8;
      *(bf16x8*)&As[r][col] = *(const bf16x8*)&A[(size_t)(bm + r) * K + k0 + col];
      *(bf16x8*)&Bs[r][col] = *(const bf16x8*)&B[(size_t)(bn + r) * K + k0 + col];
    }
    __syncthreads();
    bf16x16 af[4], bfr[2];
    #pragma unroll
    for (int i = 0; i < 4; ++i) af[i] = frag_a_ld(&As[wm * 64 + i * 16 + nl][0], half);
    #pragma unroll
    for (int j = 0; j < 2; ++j) bfr[j] = frag_b_ld(&Bs[wn * 32 + j * 16 + nl][0], half);
    #pragma unroll
    for (int i = 0; i < 4; ++i)
      #pragma unroll
      for (int j = 0; j < 2; ++j) acc[i][j] = wmma_bf16(af[i], bfr[j], acc[i][j]);
  }
#endif

  #pragma unroll
  for (int i = 0; i < 4; ++i)
    #pragma unroll
    for (int j = 0; j < 2; ++j) {
      const int col = bn + wn * 32 + j * 16 + nl;
      #pragma unroll
      for (int r = 0; r < 8; ++r) {
        const int row = bm + wm * 64 + i * 16 + r + 8 * half;
        const size_t idx = (size_t)row * N + col;
        const float v = acc[i][j][r];
        if (EPI == 0) {
          ((__bf16*)Cout)[idx] = (__bf16)v;
        } else if (EPI == 1) {
          const float g = 0.5f * v * (1.0f + erff(v * 0.70710678118654752f));
          ((__bf16*)Cout)[idx] = (__bf16)g;
        } else {
          ((float*)Cout)[idx] = v + resid[idx];
        }
      }
    }
}

// ---------------------------------------------------------------------------
// Flash attention. Q/K/V/O bf16 in [b, s, h*64 + e] layout (row stride DM).
// Block = 256 threads (8 waves): wave w owns 16 query rows; block = 128 q rows.
// 32-key tiles through LDS; online softmax; P bounced through LDS to convert
// C-layout -> A-fragment layout for the P*V WMMA.
// Grid: (SEQLEN/128, NBATCH*NHEADS)
// ---------------------------------------------------------------------------
__global__ __launch_bounds__(256) void attn_kernel(
    const __bf16* __restrict__ Q, const __bf16* __restrict__ Kt,
    const __bf16* __restrict__ Vt_g, __bf16* __restrict__ O) {
  const int bh = blockIdx.y;
  const int b  = bh >> 4;
  const int h  = bh & 15;
  const int q0 = blockIdx.x * 128;
  const int tid  = threadIdx.x;
  const int lane = tid & 31;
  const int wave = tid >> 5;
  const int half = lane >> 4;
  const int nl   = lane & 15;

  const __bf16* qp = Q    + (size_t)b * SEQLEN * DM + h * DHEAD;
  const __bf16* kp = Kt   + (size_t)b * SEQLEN * DM + h * DHEAD;
  const __bf16* vp = Vt_g + (size_t)b * SEQLEN * DM + h * DHEAD;
  __bf16*       op = O    + (size_t)b * SEQLEN * DM + h * DHEAD;

  __shared__ __bf16 Ks[32][72];      // [key][e]
  __shared__ __bf16 Vs[64][40];      // transposed: [e][key]
  __shared__ __bf16 Ps[8][16][40];   // per-wave P scratch [qrow][key]

  const __bf16* qrow = qp + (size_t)(q0 + wave * 16 + nl) * DM;
  bf16x16 qf0 = frag_a_ld(qrow + 0,  half);
  bf16x16 qf1 = frag_a_ld(qrow + 32, half);

  f32x8 o0 = vzero8(), o1 = vzero8(), o2 = vzero8(), o3 = vzero8();
  float mrun[8], lrun[8];
  #pragma unroll
  for (int r = 0; r < 8; ++r) { mrun[r] = -3.0e38f; lrun[r] = 0.f; }

  for (int kt = 0; kt < SEQLEN; kt += 32) {
    __syncthreads();
    {  // cooperative K / V(transposed) tile load: 32x64 each, 8 elems/thread
      const int r   = tid >> 3;
      const int col = (tid & 7) * 8;
      *(bf16x8*)&Ks[r][col] = *(const bf16x8*)&kp[(size_t)(kt + r) * DM + col];
      bf16x8 vv = *(const bf16x8*)&vp[(size_t)(kt + r) * DM + col];
      #pragma unroll
      for (int t = 0; t < 8; ++t) Vs[col + t][r] = vv[t];
    }
    __syncthreads();

    // scores S = Q(16x64) * K_tile^T (32 keys): two 16-key C fragments
    f32x8 sc0 = vzero8(), sc1 = vzero8();
    {
      const __bf16* k0r = &Ks[nl][0];
      const __bf16* k1r = &Ks[16 + nl][0];
      sc0 = wmma_bf16(qf0, frag_b_ld(k0r + 0,  half), sc0);
      sc0 = wmma_bf16(qf1, frag_b_ld(k0r + 32, half), sc0);
      sc1 = wmma_bf16(qf0, frag_b_ld(k1r + 0,  half), sc1);
      sc1 = wmma_bf16(qf1, frag_b_ld(k1r + 32, half), sc1);
    }

    float mt[8], rs[8];
    #pragma unroll
    for (int r = 0; r < 8; ++r) {
      sc0[r] *= 0.125f;               // 1/sqrt(64)
      sc1[r] *= 0.125f;
      mt[r] = fmaxf(sc0[r], sc1[r]);
    }
    #pragma unroll
    for (int off = 1; off < 16; off <<= 1)
      #pragma unroll
      for (int r = 0; r < 8; ++r)
        mt[r] = fmaxf(mt[r], __shfl_xor(mt[r], off, 32));
    #pragma unroll
    for (int r = 0; r < 8; ++r) {
      const float mn = fmaxf(mrun[r], mt[r]);
      const float al = __expf(mrun[r] - mn);
      mrun[r] = mn;
      sc0[r] = __expf(sc0[r] - mn);
      sc1[r] = __expf(sc1[r] - mn);
      rs[r]  = sc0[r] + sc1[r];
      o0[r] *= al; o1[r] *= al; o2[r] *= al; o3[r] *= al;
      lrun[r] *= al;
    }
    #pragma unroll
    for (int off = 1; off < 16; off <<= 1)
      #pragma unroll
      for (int r = 0; r < 8; ++r)
        rs[r] += __shfl_xor(rs[r], off, 32);
    #pragma unroll
    for (int r = 0; r < 8; ++r) {
      lrun[r] += rs[r];
      Ps[wave][r + 8 * half][nl]      = (__bf16)sc0[r];  // C-layout -> LDS
      Ps[wave][r + 8 * half][16 + nl] = (__bf16)sc1[r];
    }
    __syncthreads();

    // O += P(16x32) * V_tile(32x64)
    {
      bf16x16 pf = frag_a_ld(&Ps[wave][nl][0], half);
      o0 = wmma_bf16(pf, frag_b_ld(&Vs[ 0 + nl][0], half), o0);
      o1 = wmma_bf16(pf, frag_b_ld(&Vs[16 + nl][0], half), o1);
      o2 = wmma_bf16(pf, frag_b_ld(&Vs[32 + nl][0], half), o2);
      o3 = wmma_bf16(pf, frag_b_ld(&Vs[48 + nl][0], half), o3);
    }
  }

  #pragma unroll
  for (int r = 0; r < 8; ++r) {
    const float invl = 1.0f / lrun[r];
    const int row = q0 + wave * 16 + r + 8 * half;
    __bf16* orow = op + (size_t)row * DM;
    orow[ 0 + nl] = (__bf16)(o0[r] * invl);
    orow[16 + nl] = (__bf16)(o1[r] * invl);
    orow[32 + nl] = (__bf16)(o2[r] * invl);
    orow[48 + nl] = (__bf16)(o3[r] * invl);
  }
}

// ---------------------------------------------------------------------------
// Host-side orchestration
// ---------------------------------------------------------------------------
extern "C" void kernel_launch(void* const* d_in, const int* in_sizes, int n_in,
                              void* d_out, int out_size, void* d_ws, size_t ws_size,
                              hipStream_t stream) {
  (void)in_sizes; (void)n_in; (void)out_size; (void)ws_size;
  const float* x  = (const float*)d_in[0];
  const float* wq = (const float*)d_in[1];
  const float* wk = (const float*)d_in[2];
  const float* wv = (const float*)d_in[3];
  const float* wo = (const float*)d_in[4];
  const float* w1 = (const float*)d_in[5];
  const float* w2 = (const float*)d_in[6];
  const float* g1 = (const float*)d_in[7];
  const float* b1 = (const float*)d_in[8];
  const float* g2 = (const float*)d_in[9];
  const float* b2 = (const float*)d_in[10];

  char* ws = (char*)d_ws;
  auto alloc = [&](size_t bytes) -> char* {
    char* p = ws;
    ws += (bytes + 255) & ~(size_t)255;
    return p;
  };
  __bf16* h_bf   = (__bf16*)alloc((size_t)ROWS * DM * 2);
  __bf16* wq_bf  = (__bf16*)alloc((size_t)DM * DM * 2);
  __bf16* wk_bf  = (__bf16*)alloc((size_t)DM * DM * 2);
  __bf16* wv_bf  = (__bf16*)alloc((size_t)DM * DM * 2);
  __bf16* wo_bf  = (__bf16*)alloc((size_t)DM * DM * 2);
  __bf16* w1_bf  = (__bf16*)alloc((size_t)DFF * DM * 2);
  __bf16* w2_bf  = (__bf16*)alloc((size_t)DM * DFF * 2);
  __bf16* q_bf   = (__bf16*)alloc((size_t)ROWS * DM * 2);
  __bf16* k_bf   = (__bf16*)alloc((size_t)ROWS * DM * 2);
  __bf16* v_bf   = (__bf16*)alloc((size_t)ROWS * DM * 2);
  __bf16* ctx_bf = (__bf16*)alloc((size_t)ROWS * DM * 2);
  float*  y1     = (float*)alloc((size_t)ROWS * DM * 4);
  __bf16* h2_bf  = (__bf16*)alloc((size_t)ROWS * DM * 2);
  __bf16* ff1_bf = (__bf16*)alloc((size_t)ROWS * DFF * 2);

  auto cvt = [&](const float* src, __bf16* dst, int n) {
    cvt_f32_bf16<<<dim3((n + 255) / 256), dim3(256), 0, stream>>>(src, dst, n);
  };
  cvt(wq, wq_bf, DM * DM);
  cvt(wk, wk_bf, DM * DM);
  cvt(wv, wv_bf, DM * DM);
  cvt(wo, wo_bf, DM * DM);
  cvt(w1, w1_bf, DFF * DM);
  cvt(w2, w2_bf, DM * DFF);

  // h = LN1(x)  (bf16)
  layernorm_to_bf16<<<dim3(ROWS), dim3(256), 0, stream>>>(x, g1, b1, h_bf);

  // Q/K/V projections: [4096,1024] x [1024,1024]^T
  dim3 blk(256);
  dim3 gproj(ROWS / 128, DM / 128);
  gemm_bf16_nt<0><<<gproj, blk, 0, stream>>>(h_bf, wq_bf, nullptr, q_bf, ROWS, DM, DM);
  gemm_bf16_nt<0><<<gproj, blk, 0, stream>>>(h_bf, wk_bf, nullptr, k_bf, ROWS, DM, DM);
  gemm_bf16_nt<0><<<gproj, blk, 0, stream>>>(h_bf, wv_bf, nullptr, v_bf, ROWS, DM, DM);

  // flash attention -> ctx (bf16, [b,s,h*64+e])
  attn_kernel<<<dim3(SEQLEN / 128, NBATCH * NHEADS), blk, 0, stream>>>(q_bf, k_bf, v_bf,
                                                                       ctx_bf);

  // y1 = x + ctx @ wo^T  (f32)
  gemm_bf16_nt<2><<<gproj, blk, 0, stream>>>(ctx_bf, wo_bf, x, y1, ROWS, DM, DM);

  // h2 = LN2(y1)
  layernorm_to_bf16<<<dim3(ROWS), dim3(256), 0, stream>>>(y1, g2, b2, h2_bf);

  // ff1 = gelu(h2 @ w1^T)  (bf16, [4096,4096])
  dim3 gff1(ROWS / 128, DFF / 128);
  gemm_bf16_nt<1><<<gff1, blk, 0, stream>>>(h2_bf, w1_bf, nullptr, ff1_bf, ROWS, DFF, DM);

  // out = x + ff1 @ w2^T  (f32)
  gemm_bf16_nt<2><<<gproj, blk, 0, stream>>>(ff1_bf, w2_bf, x, (float*)d_out, ROWS, DM,
                                             DFF);
}
